// SFFN_57028575756706
// MI455X (gfx1250) — matrix operand order repaired and checked by reading, not compiled, not run on hardware
//
#include <hip/hip_runtime.h>
#include <math.h>

typedef __attribute__((ext_vector_type(16))) __bf16 v16bf;
typedef __attribute__((ext_vector_type(8)))  __bf16 bf16x8;
typedef __attribute__((ext_vector_type(8)))  float  v8f;

#define B_DIM   4
#define T_DIM   4096
#define D_DIM   2048
#define LATENT  256
#define FFN_DIM 1024
#define TEMB    2048

// ---------------------------------------------------------------------------
// WMMA fragment helpers (bf16, 16x16x32). Layouts per CDNA5 ISA 7.12.2:
//  A (16x32): lane L holds row L%16; halves K=[(L/16)*8,+8) and [16+(L/16)*8,+8)
//  B (32x16): lane L holds col N=L%16; same K halves as A, read from the
//             weight matrix stored N-major (so we pre-transpose weights).
//  C/D (16x16 f32): element j of v8f in lane L is (M = j + 8*(L/16), N = L%16)
// ---------------------------------------------------------------------------
__device__ __forceinline__ v16bf frag_a(const __bf16* base, int stride, int lane) {
  const int hq = lane >> 4;
  const int r  = lane & 15;
  const __bf16* p = base + (size_t)r * stride;
  union { v16bf v; bf16x8 h[2]; } u;
  u.h[0] = *(const bf16x8*)(p + hq * 8);
  u.h[1] = *(const bf16x8*)(p + 16 + hq * 8);
  return u.v;
}

__device__ __forceinline__ v16bf frag_b(const __bf16* base, int stride, int lane) {
  const int hq = lane >> 4;
  const int n  = lane & 15;
  const __bf16* p = base + (size_t)n * stride;
  union { v16bf v; bf16x8 h[2]; } u;
  u.h[0] = *(const bf16x8*)(p + hq * 8);
  u.h[1] = *(const bf16x8*)(p + 16 + hq * 8);
  return u.v;
}

__device__ __forceinline__ v8f wmma_bf16(v16bf a, v16bf b, v8f c) {
  // (neg_a, A, neg_b, B, c_mod, C, reuse_a, reuse_b)
  return __builtin_amdgcn_wmma_f32_16x16x32_bf16(false, a, false, b, (short)0, c,
                                                 false, false);
}

// ---------------------------------------------------------------------------
// K0: transpose + convert f32 (K,N) row-major -> bf16 (N,K) row-major, per
// expert slice selected by blockIdx.z.
// ---------------------------------------------------------------------------
__global__ __launch_bounds__(256) void convT_kernel(const float* __restrict__ in,
                                                    __bf16* __restrict__ outp,
                                                    int K, int N) {
  const size_t total = (size_t)K * N;
  const float* src = in  + (size_t)blockIdx.z * total;
  __bf16*      dst = outp + (size_t)blockIdx.z * total;
  for (size_t idx = (size_t)blockIdx.x * 256 + threadIdx.x; idx < total;
       idx += (size_t)gridDim.x * 256) {
    const int n = (int)(idx / (size_t)K);
    const int k = (int)(idx % (size_t)K);
    dst[idx] = (__bf16)src[(size_t)k * N + n];
  }
}

// ---------------------------------------------------------------------------
// K1: eo[b, 0:2D] = silu(emb[b]) @ emb_W + emb_b      (tiny: 4 x 4096)
// ---------------------------------------------------------------------------
__global__ __launch_bounds__(256) void emb_kernel(const float* __restrict__ emb,
                                                  const float* __restrict__ emb_W,
                                                  const float* __restrict__ emb_b,
                                                  float* __restrict__ eo) {
  __shared__ float se[TEMB];
  const int b   = blockIdx.y;
  const int col = blockIdx.x * 256 + threadIdx.x;
#pragma unroll
  for (int i = 0; i < TEMB / 256; ++i) {
    const int k = threadIdx.x + i * 256;
    const float e = emb[b * TEMB + k];
    se[k] = e / (1.f + expf(-e));
  }
  __syncthreads();
  float s = emb_b[col];
  for (int k = 0; k < TEMB; ++k)
    s += se[k] * emb_W[(size_t)k * (2 * D_DIM) + col];
  eo[(size_t)b * 2 * D_DIM + col] = s;
}

// ---------------------------------------------------------------------------
// K2: per (32-token tile, expert): y = GELU(x_e @ W1_e + b1_e) @ W2_e + b2_e
// 8 waves / 256 threads. Dynamic LDS: x tile (16KB) + h tile (64KB) bf16.
// M-tile of 32 rows: each B fragment feeds 2 WMMAs -> half the weight traffic.
// ---------------------------------------------------------------------------
__global__ __launch_bounds__(256) void ffn_kernel(const float* __restrict__ x,
                                                  const float* __restrict__ b1,
                                                  const float* __restrict__ b2,
                                                  const __bf16* __restrict__ W1bf,
                                                  const __bf16* __restrict__ W2bf,
                                                  __bf16* __restrict__ ybf) {
  extern __shared__ char smem_f[];
  __bf16* Ax = (__bf16*)smem_f;                          // 32 x 256  (16 KB)
  __bf16* Hs = (__bf16*)(smem_f + 32 * LATENT * 2);      // 32 x 1024 (64 KB)

  const int m0  = blockIdx.x * 32;
  const int e   = blockIdx.y;
  const int tid = threadIdx.x;

  // stage x tile (32 rows x 256 latent cols) as bf16
  {
    const int r  = tid >> 3;
    const int c0 = (tid & 7) * 32;
    const float* src = x + (size_t)(m0 + r) * D_DIM + e * LATENT + c0;
    __bf16* dst = Ax + r * LATENT + c0;
#pragma unroll
    for (int i = 0; i < 32; i += 4) {
      const float4 v = *(const float4*)(src + i);
      dst[i]     = (__bf16)v.x;
      dst[i + 1] = (__bf16)v.y;
      dst[i + 2] = (__bf16)v.z;
      dst[i + 3] = (__bf16)v.w;
    }
  }
  __syncthreads();

  const int wave = tid >> 5;
  const int lane = tid & 31;
  const int hq   = lane >> 4;
  const int rl   = lane & 15;
  const v8f vzero = {0.f, 0.f, 0.f, 0.f, 0.f, 0.f, 0.f, 0.f};

  // ---- h = x @ W1_e : wave owns 128 of 1024 ffn cols, 2 M-subtiles ----
  const __bf16* W1e = W1bf + (size_t)e * FFN_DIM * LATENT;  // (1024 N, 256 K)
  v8f acc[2][8];
#pragma unroll
  for (int mt = 0; mt < 2; ++mt)
#pragma unroll
    for (int nt = 0; nt < 8; ++nt) acc[mt][nt] = vzero;

  for (int kt = 0; kt < LATENT / 32; ++kt) {
    v16bf a0 = frag_a(Ax + kt * 32, LATENT, lane);
    v16bf a1 = frag_a(Ax + 16 * LATENT + kt * 32, LATENT, lane);
#pragma unroll
    for (int nt = 0; nt < 8; ++nt) {
      const int n0 = wave * 128 + nt * 16;
      const v16bf b = frag_b(W1e + (size_t)n0 * LATENT + kt * 32, LATENT, lane);
      acc[0][nt] = wmma_bf16(a0, b, acc[0][nt]);
      acc[1][nt] = wmma_bf16(a1, b, acc[1][nt]);
    }
  }
  // bias + exact GELU, write to LDS
#pragma unroll
  for (int nt = 0; nt < 8; ++nt) {
    const int n = wave * 128 + nt * 16 + rl;
    const float bias = b1[e * FFN_DIM + n];
#pragma unroll
    for (int mt = 0; mt < 2; ++mt) {
#pragma unroll
      for (int j = 0; j < 8; ++j) {
        const int M = mt * 16 + j + 8 * hq;
        float v = acc[mt][nt][j] + bias;
        v = 0.5f * v * (1.0f + erff(v * 0.70710678118654752f));
        Hs[M * FFN_DIM + n] = (__bf16)v;
      }
    }
  }
  __syncthreads();

  // ---- y = h @ W2_e : wave owns 32 of 256 latent cols, 2 M-subtiles ----
  const __bf16* W2e = W2bf + (size_t)e * LATENT * FFN_DIM;  // (256 N, 1024 K)
  v8f acc2[2][2];
#pragma unroll
  for (int mt = 0; mt < 2; ++mt) {
    acc2[mt][0] = vzero;
    acc2[mt][1] = vzero;
  }
  for (int kt = 0; kt < FFN_DIM / 32; ++kt) {
    v16bf a0 = frag_a(Hs + kt * 32, FFN_DIM, lane);
    v16bf a1 = frag_a(Hs + 16 * FFN_DIM + kt * 32, FFN_DIM, lane);
#pragma unroll
    for (int nt = 0; nt < 2; ++nt) {
      const int n0 = wave * 32 + nt * 16;
      const v16bf b = frag_b(W2e + (size_t)n0 * FFN_DIM + kt * 32, FFN_DIM, lane);
      acc2[0][nt] = wmma_bf16(a0, b, acc2[0][nt]);
      acc2[1][nt] = wmma_bf16(a1, b, acc2[1][nt]);
    }
  }
#pragma unroll
  for (int nt = 0; nt < 2; ++nt) {
    const int n = wave * 32 + nt * 16 + rl;
    const float bias = b2[e * LATENT + n];
#pragma unroll
    for (int mt = 0; mt < 2; ++mt) {
#pragma unroll
      for (int j = 0; j < 8; ++j) {
        const int M = m0 + mt * 16 + j + 8 * hq;
        ybf[(size_t)M * D_DIM + e * LATENT + n] = (__bf16)(acc2[mt][nt][j] + bias);
      }
    }
  }
}

// ---------------------------------------------------------------------------
// K3: per (32-token tile, half of N): LN(y)*(1+scale)+shift, SiLU, @out_W,
// + out_b + residual x.  Dynamic LDS: 32x2048 bf16 A tile (128KB) + stats.
// ---------------------------------------------------------------------------
__global__ __launch_bounds__(256) void out_kernel(const __bf16* __restrict__ ybf,
                                                  const float* __restrict__ eo,
                                                  const float* __restrict__ ln_g,
                                                  const float* __restrict__ ln_b,
                                                  const __bf16* __restrict__ outWbf,
                                                  const float* __restrict__ out_b,
                                                  const float* __restrict__ x,
                                                  float* __restrict__ out) {
  extern __shared__ char smem[];
  __bf16* As  = (__bf16*)smem;                       // 32 x 2048 bf16 (128 KB)
  float* ssum = (float*)(smem + 32 * D_DIM * 2);     // [32]
  float* ssq  = ssum + 32;                           // [32]

  const int m0  = blockIdx.x * 32;
  const int nh  = blockIdx.y;
  const int bb  = m0 / T_DIM;   // batch index (T divisible by 32)
  const int tid = threadIdx.x;

  if (tid < 32) { ssum[tid] = 0.f; ssq[tid] = 0.f; }
  __syncthreads();

  // stage y tile into LDS + per-row partial LN sums (ds_add_f32 atomics)
  {
    const int r  = tid >> 3;
    const int c0 = (tid & 7) * 256;
    const __bf16* src = ybf + (size_t)(m0 + r) * D_DIM + c0;
    __bf16* dst = As + r * D_DIM + c0;
    float s = 0.f, sq = 0.f;
#pragma unroll 4
    for (int i = 0; i < 256; i += 8) {
      const bf16x8 v = *(const bf16x8*)(src + i);
      *(bf16x8*)(dst + i) = v;
#pragma unroll
      for (int j = 0; j < 8; ++j) {
        const float f = (float)v[j];
        s  += f;
        sq += f * f;
      }
    }
    atomicAdd(&ssum[r], s);
    atomicAdd(&ssq[r], sq);
  }
  __syncthreads();

  // transform in place: silu( (LN(y)*g+b) * (1+scale) + shift )
  {
    const int r  = tid >> 3;
    const int c0 = (tid & 7) * 256;
    const float mu   = ssum[r] * (1.f / D_DIM);
    const float var  = ssq[r] * (1.f / D_DIM) - mu * mu;
    const float rstd = rsqrtf(var + 1e-5f);
    __bf16* dst = As + r * D_DIM + c0;
    const float* gp  = ln_g + c0;
    const float* bp  = ln_b + c0;
    const float* scp = eo + (size_t)bb * 2 * D_DIM + c0;
    const float* shp = scp + D_DIM;
#pragma unroll 4
    for (int i = 0; i < 256; ++i) {
      const float v  = (float)dst[i];
      const float ln = (v - mu) * rstd * gp[i] + bp[i];
      const float hn = ln * (1.f + scp[i]) + shp[i];
      dst[i] = (__bf16)(hn / (1.f + expf(-hn)));
    }
  }
  __syncthreads();

  // GEMM: out tile (32 x 1024 half) = As @ outWbf (N-major), K = 2048
  const int wave = tid >> 5;
  const int lane = tid & 31;
  const int hq   = lane >> 4;
  const int rl   = lane & 15;
  const int n0w  = nh * 1024 + wave * 128;
  const v8f vzero = {0.f, 0.f, 0.f, 0.f, 0.f, 0.f, 0.f, 0.f};
  v8f acc[2][8];
#pragma unroll
  for (int mt = 0; mt < 2; ++mt)
#pragma unroll
    for (int nt = 0; nt < 8; ++nt) acc[mt][nt] = vzero;

  for (int kt = 0; kt < D_DIM / 32; ++kt) {
    if (kt + 1 < D_DIM / 32)
      __builtin_prefetch(outWbf + (size_t)n0w * D_DIM + (kt + 1) * 32, 0, 1);
    const v16bf a0 = frag_a(As + kt * 32, D_DIM, lane);
    const v16bf a1 = frag_a(As + 16 * D_DIM + kt * 32, D_DIM, lane);
#pragma unroll
    for (int nt = 0; nt < 8; ++nt) {
      const v16bf b =
          frag_b(outWbf + (size_t)(n0w + nt * 16) * D_DIM + kt * 32, D_DIM, lane);
      acc[0][nt] = wmma_bf16(a0, b, acc[0][nt]);
      acc[1][nt] = wmma_bf16(a1, b, acc[1][nt]);
    }
  }

  // epilogue: + out_b + residual x
#pragma unroll
  for (int nt = 0; nt < 8; ++nt) {
    const int n = n0w + nt * 16 + rl;
    const float bias = out_b[n];
#pragma unroll
    for (int mt = 0; mt < 2; ++mt) {
#pragma unroll
      for (int j = 0; j < 8; ++j) {
        const int M = m0 + mt * 16 + j + 8 * hq;
        out[(size_t)M * D_DIM + n] = x[(size_t)M * D_DIM + n] + acc[mt][nt][j] + bias;
      }
    }
  }
}

// ---------------------------------------------------------------------------
extern "C" void kernel_launch(void* const* d_in, const int* in_sizes, int n_in,
                              void* d_out, int out_size, void* d_ws, size_t ws_size,
                              hipStream_t stream) {
  (void)in_sizes; (void)n_in; (void)out_size; (void)ws_size;
  const float* x     = (const float*)d_in[0];
  const float* emb   = (const float*)d_in[1];
  const float* W1    = (const float*)d_in[2];
  const float* b1    = (const float*)d_in[3];
  const float* W2    = (const float*)d_in[4];
  const float* b2    = (const float*)d_in[5];
  const float* emb_W = (const float*)d_in[6];
  const float* emb_b = (const float*)d_in[7];
  const float* ln_g  = (const float*)d_in[8];
  const float* ln_b  = (const float*)d_in[9];
  const float* out_W = (const float*)d_in[10];
  const float* out_b = (const float*)d_in[11];
  float* out = (float*)d_out;

  char* ws = (char*)d_ws;
  size_t off = 0;
  float* eo = (float*)(ws + off);
  off += (size_t)B_DIM * 2 * D_DIM * sizeof(float);            // 64 KB
  __bf16* W1bf = (__bf16*)(ws + off);
  off += (size_t)8 * FFN_DIM * LATENT * sizeof(__bf16);        // 4 MB
  __bf16* W2bf = (__bf16*)(ws + off);
  off += (size_t)8 * LATENT * FFN_DIM * sizeof(__bf16);        // 4 MB
  __bf16* outWbf = (__bf16*)(ws + off);
  off += (size_t)D_DIM * D_DIM * sizeof(__bf16);               // 8 MB
  __bf16* ybf = (__bf16*)(ws + off);                           // 64 MB

  // one-time-per-call weight transpose + bf16 convert
  convT_kernel<<<dim3(1024, 1, 8), 256, 0, stream>>>(W1, W1bf, LATENT, FFN_DIM);
  convT_kernel<<<dim3(1024, 1, 8), 256, 0, stream>>>(W2, W2bf, FFN_DIM, LATENT);
  convT_kernel<<<dim3(4096, 1, 1), 256, 0, stream>>>(out_W, outWbf, D_DIM, D_DIM);

  // time-embedding scale/shift
  emb_kernel<<<dim3((2 * D_DIM) / 256, B_DIM), 256, 0, stream>>>(emb, emb_W, emb_b, eo);

  // expert FFN (WMMA bf16) -> ybf   (32-token tiles)
  const size_t smem_ffn = (size_t)32 * LATENT * 2 + (size_t)32 * FFN_DIM * 2;  // 80 KB
  ffn_kernel<<<dim3((B_DIM * T_DIM) / 32, 8), 256, smem_ffn, stream>>>(
      x, b1, b2, W1bf, W2bf, ybf);

  // LN + stylization + SiLU + out projection + residual (WMMA bf16)
  const size_t smem_out = (size_t)32 * D_DIM * 2 + 64 * sizeof(float);         // 128.25 KB
  out_kernel<<<dim3((B_DIM * T_DIM) / 32, 2), 256, smem_out, stream>>>(
      ybf, eo, ln_g, ln_b, outWbf, out_b, x, out);
}
